// Cvx_WarcraftShortestPathNet_49589692400117
// MI455X (gfx1250) — compile-verified, matrix operand order checked by reading, not run on hardware
//
#include <hip/hip_runtime.h>
#include <hip/hip_bf16.h>
#include <stdint.h>
#include <stddef.h>

// ---------------------------------------------------------------------------
// CDNA5 (gfx1250) WMMA helpers — wave32, V_WMMA_F32_16X16X32_BF16
// ---------------------------------------------------------------------------
typedef __attribute__((ext_vector_type(16))) __bf16 v16bf;
typedef __attribute__((ext_vector_type(8)))  float  v8f;

union FragBF { v16bf v; __bf16 e[16]; };
union FragF  { v8f   v; float  e[8];  };

// ISA 7.12.2: 16-bit A-matrix 16x32 layout. For lane L (hi = L>=16) and
// bf16 slot s (s=2*i+h, i=VGPR index, h=low/high half):
//   i<4 : K = 2*i      + (hi?8:0) + h
//   i>=4: K = 2*i + 8  + (hi?8:0) + h
__device__ __forceinline__ int akidx(int s, bool hi) {
  int i = s >> 1, h = s & 1;
  int base = (i < 4) ? (2 * i) : (2 * i + 8);
  return base + (hi ? 8 : 0) + h;
}

// B fragment (32x16, K-major per lane) is the A-layout of B^T:
// lane L holds column n = L&15, same K mapping as above.

__device__ __forceinline__ v8f wmma_bf16(v16bf a, v16bf b, v8f c) {
  // 8 args: (neg_a, A, neg_b, B, c_mod, C, reuse_a, reuse_b)
  return __builtin_amdgcn_wmma_f32_16x16x32_bf16(false, a, false, b,
                                                 (short)0, c, false, false);
}

// ---------------------------------------------------------------------------
// Tensor Data Mover (TDM): async DMA of a 2D bf16 tile (16 cols x `rows` rows,
// row stride `strideElems`) from global into LDS. Descriptor per ISA ch.8 D#.
// ---------------------------------------------------------------------------
#if defined(__HIP_DEVICE_COMPILE__) && __has_builtin(__builtin_amdgcn_tensor_load_to_lds)
#define HAVE_TDM 1
typedef __attribute__((ext_vector_type(4))) unsigned int u32x4;
typedef __attribute__((ext_vector_type(4))) int          i32x4;
typedef __attribute__((ext_vector_type(8))) int          i32x8;

__device__ __forceinline__ void tdm_load_bf16_tile(uint32_t lds_addr, uint64_t gaddr,
                                                   int rows, int strideElems)
{
  // D# group 0: count=1 | lds_addr | global_addr[56:0] | type=2 (bits 127:126)
  u32x4 g0;
  g0[0] = 1u;                                                // valid user D#
  g0[1] = lds_addr;                                          // LDS byte address
  g0[2] = (uint32_t)gaddr;                                   // global addr lo
  g0[3] = ((uint32_t)(gaddr >> 32) & 0x01FFFFFFu) | 0x80000000u;
  // D# group 1: data_size=1 (2B); tensor 16 x rows, stride strideElems;
  // tile 16 x rows; no multicast/pad/iterate/atomic-barrier.
  i32x8 g1;
  g1[0] = 0x00010000;                 // [17:16] data_size = 1 (2 bytes)
  g1[1] = (16 & 0xFFFF) << 16;        // tensor_dim0[15:0]  (bits 63:48)
  g1[2] = (rows & 0xFFFF) << 16;      // tensor_dim1[15:0]  (bits 95:80)
  g1[3] = (16 & 0xFFFF) << 16;        // tile_dim0          (bits 127:112)
  g1[4] = rows & 0xFFFF;              // tile_dim1          (bits 143:128)
  g1[5] = strideElems;                // tensor_dim0_stride[31:0]
  g1[6] = 0;
  g1[7] = 0;
  i32x4 gz = {0, 0, 0, 0};            // groups 2/3 unused (2D tensor)
#if __clang_major__ >= 23
  i32x8 gz8 = {0, 0, 0, 0, 0, 0, 0, 0};
  __builtin_amdgcn_tensor_load_to_lds(g0, g1, gz, gz, gz8, 0);
#else
  __builtin_amdgcn_tensor_load_to_lds(g0, g1, gz, gz, 0);
#endif
}

__device__ __forceinline__ void tdm_wait()
{
#if __has_builtin(__builtin_amdgcn_s_wait_tensorcnt)
  __builtin_amdgcn_s_wait_tensorcnt(0);
#else
  asm volatile("s_wait_tensorcnt 0x0");
#endif
  asm volatile("" ::: "memory");      // keep LDS reads below the wait
}
#else
#define HAVE_TDM 0
#endif

// ---------------------------------------------------------------------------
// Implicit-GEMM convolution (NCHW / OIHW), bf16 WMMA with f32 accumulate.
// One wave computes a 16(oc) x 16(pixel) tile; K = Cin*KH*KW padded to 32.
// ---------------------------------------------------------------------------
__global__ void k_conv_wmma(const float* __restrict__ in,
                            const float* __restrict__ wgt,
                            float* __restrict__ out,
                            int N, int Cin, int H, int W, int Cout,
                            int KH, int KW, int stride, int pad,
                            int OH, int OW)
{
  const int Ktot = Cin * KH * KW;
  const int numN = (N * OH * OW) >> 4;   // pixel tiles (sizes chosen divisible)
  const int numTiles = (Cout >> 4) * numN;
  int wave = blockIdx.x * (blockDim.x >> 5) + (threadIdx.x >> 5);
  const bool active = wave < numTiles;
  if (!active) wave = 0;
  const int mt = wave / numN, nt = wave % numN;

  const int L  = threadIdx.x & 31;
  const bool hi = L >= 16;
  const int lm = L & 15;

  const int oc = mt * 16 + lm;                 // A-fragment row (weights)
  const int px = nt * 16 + lm;                 // B-fragment column (pixel)
  const int ow = px % OW; int t = px / OW;
  const int oh = t % OH;  const int bn = t / OH;
  const int ihb = oh * stride - pad, iwb = ow * stride - pad;

  const float* inb  = in  + (size_t)bn * Cin * H * W;
  const float* wrow = wgt + (size_t)oc * Ktot;

  FragF acc;
#pragma unroll
  for (int r = 0; r < 8; ++r) acc.e[r] = 0.f;

  const int ksteps = (Ktot + 31) >> 5;
  for (int ks = 0; ks < ksteps; ++ks) {
    const int kb = ks << 5;
    FragBF fa, fb;
#pragma unroll
    for (int s = 0; s < 16; ++s) {
      const int k = kb + akidx(s, hi);
      float av = 0.f, bv = 0.f;
      if (k < Ktot) {
        av = wrow[k];
        const int kw = k % KW; const int t2 = k / KW;
        const int kh = t2 % KH; const int ci = t2 / KH;
        const int ih = ihb + kh, iw = iwb + kw;
        if (ih >= 0 && ih < H && iw >= 0 && iw < W)
          bv = inb[((size_t)ci * H + ih) * W + iw];
      }
      fa.e[s] = (__bf16)av;
      fb.e[s] = (__bf16)bv;
    }
    acc.v = wmma_bf16(fa.v, fb.v, acc.v);
  }

  if (active) {
#pragma unroll
    for (int r = 0; r < 8; ++r) {
      const int oc2 = mt * 16 + r + (hi ? 8 : 0);  // C layout: m = r (+8 hi lanes)
      out[(((size_t)bn * Cout + oc2) * OH + oh) * OW + ow] = acc.e[r];
    }
  }
}

// ---------------------------------------------------------------------------
// BatchNorm (training stats): one block per channel, LDS tree reduction.
// ---------------------------------------------------------------------------
__global__ void k_bn_stats(const float* __restrict__ x, float* __restrict__ mu,
                           float* __restrict__ rs, int N, int C, int HW)
{
  __shared__ float s1[256], s2[256];
  const int c = blockIdx.x;
  const int total = N * HW;
  float sum = 0.f, sq = 0.f;
  for (int i = threadIdx.x; i < total; i += blockDim.x) {
    const int n = i / HW, p = i - n * HW;
    const float v = x[((size_t)n * C + c) * HW + p];
    sum += v; sq += v * v;
  }
  s1[threadIdx.x] = sum; s2[threadIdx.x] = sq;
  __syncthreads();
  for (int o = 128; o > 0; o >>= 1) {
    if (threadIdx.x < o) { s1[threadIdx.x] += s1[threadIdx.x + o];
                           s2[threadIdx.x] += s2[threadIdx.x + o]; }
    __syncthreads();
  }
  if (threadIdx.x == 0) {
    const float m = s1[0] / total;
    const float v = s2[0] / total - m * m;
    mu[c] = m; rs[c] = rsqrtf(v + 1e-5f);
  }
}

__global__ void k_bn_apply(const float* __restrict__ x, const float* __restrict__ mu,
                           const float* __restrict__ rs, const float* __restrict__ g,
                           const float* __restrict__ bt, const float* __restrict__ res,
                           float* __restrict__ y, int total, int C, int HW, int dorelu)
{
  const int i = blockIdx.x * blockDim.x + threadIdx.x;
  if (i >= total) return;
  const int c = (i / HW) % C;
  float v = (x[i] - mu[c]) * rs[c] * g[c] + bt[c];
  if (res) v += res[i];
  if (dorelu) v = fmaxf(v, 0.f);
  y[i] = v;
}

__global__ void k_maxpool(const float* __restrict__ x, float* __restrict__ y,
                          int NC, int H, int W, int OH, int OW)
{
  const int i = blockIdx.x * blockDim.x + threadIdx.x;
  if (i >= NC * OH * OW) return;
  const int ow = i % OW; int t = i / OW; const int oh = t % OH; const int nc = t / OH;
  const float* xp = x + (size_t)nc * H * W;
  const int h0 = oh * 2 - 1, w0 = ow * 2 - 1;
  float m = -INFINITY;
  for (int dh = 0; dh < 3; ++dh) { const int h = h0 + dh; if (h < 0 || h >= H) continue;
    for (int dw = 0; dw < 3; ++dw) { const int w = w0 + dw; if (w < 0 || w >= W) continue;
      m = fmaxf(m, xp[h * W + w]); } }
  y[i] = m;
}

__global__ void k_conv1x1(const float* __restrict__ x, const float* __restrict__ w,
                          float* __restrict__ y, int N, int C, int HW)
{
  const int i = blockIdx.x * blockDim.x + threadIdx.x;
  if (i >= N * HW) return;
  const int b = i / HW, p = i - b * HW;
  const float* xp = x + (size_t)b * C * HW + p;
  float s = 0.f;
  for (int c = 0; c < C; ++c) s += xp[(size_t)c * HW] * w[c];
  y[i] = s;
}

// adaptive max pool 24->12 is exactly 2x2 windows; cost is (128,144)
__global__ void k_amax(const float* __restrict__ x, float* __restrict__ cost)
{
  const int i = blockIdx.x * blockDim.x + threadIdx.x;
  if (i >= 128 * 144) return;
  const int b = i / 144, j = i - b * 144;
  const int r = j / 12, c = j - r * 12;
  const float* xp = x + (size_t)b * 576;
  float m = -INFINITY;
  for (int dr = 0; dr < 2; ++dr)
    for (int dc = 0; dc < 2; ++dc)
      m = fmaxf(m, xp[(2 * r + dr) * 24 + (2 * c + dc)]);
  cost[i] = m;
}

// w[b,e] = cost[b, dst[e]] into 1024-padded buffer; z init to 0
__global__ void k_gather_w(const float* __restrict__ cost, const int* __restrict__ dst,
                           float* __restrict__ wpad, float* __restrict__ z, int NE)
{
  const int i = blockIdx.x * blockDim.x + threadIdx.x;
  if (i >= 128 * 1024) return;
  const int b = i >> 10, e = i & 1023;
  wpad[i] = (e < NE) ? cost[b * 144 + dst[e]] : 0.f;
  z[i] = 0.f;
}

// A (144x1012 f32, entries in {-1,0,1}: exact in bf16) -> 160x1024 zero-padded bf16
__global__ void k_prep_Abf(const float* __restrict__ A, __bf16* __restrict__ Abf)
{
  const int i = blockIdx.x * blockDim.x + threadIdx.x;
  if (i >= 160 * 1024) return;
  const int row = i >> 10, e = i & 1023;
  const float v = (row < 144 && e < 1012) ? A[row * 1012 + e] : 0.f;
  Abf[i] = (__bf16)v;
}

// M = A*A^T (144x144) via bf16 WMMA (exact: small-integer products)
__global__ void k_maat(const __bf16* __restrict__ Abf, float* __restrict__ M)
{
  const int numN = 9, numTiles = 81;
  int wave = blockIdx.x * (blockDim.x >> 5) + (threadIdx.x >> 5);
  const bool active = wave < numTiles;
  if (!active) wave = 0;
  const int mt = wave / numN, nt = wave % numN;
  const int L = threadIdx.x & 31; const bool hi = L >= 16; const int lm = L & 15;
  const __bf16* arow = Abf + (size_t)(mt * 16 + lm) * 1024;
  const __bf16* brow = Abf + (size_t)(nt * 16 + lm) * 1024;
  FragF acc;
#pragma unroll
  for (int r = 0; r < 8; ++r) acc.e[r] = 0.f;
  for (int ks = 0; ks < 32; ++ks) {
    const int kb = ks << 5;
    FragBF fa, fb;
#pragma unroll
    for (int s = 0; s < 16; ++s) {
      const int k = kb + akidx(s, hi);
      fa.e[s] = arow[k];
      fb.e[s] = brow[k];
    }
    acc.v = wmma_bf16(fa.v, fb.v, acc.v);
  }
  if (active) {
#pragma unroll
    for (int r = 0; r < 8; ++r) {
      const int row = mt * 16 + r + (hi ? 8 : 0);
      const int col = nt * 16 + lm;
      M[row * 144 + col] = acc.e[r];
    }
  }
}

// alpha = 1 / (||M||_1 * ||M||_inf) (M symmetric -> max abs row sum squared)
__global__ void k_alpha(const float* __restrict__ M, float* __restrict__ alpha)
{
  __shared__ float s[256];
  float r = 0.f;
  for (int i = threadIdx.x; i < 144; i += blockDim.x) {
    float rs = 0.f;
    for (int j = 0; j < 144; ++j) rs += fabsf(M[i * 144 + j]);
    r = fmaxf(r, rs);
  }
  s[threadIdx.x] = r; __syncthreads();
  for (int o = 128; o > 0; o >>= 1) {
    if (threadIdx.x < o) s[threadIdx.x] = fmaxf(s[threadIdx.x], s[threadIdx.x + o]);
    __syncthreads();
  }
  if (threadIdx.x == 0) alpha[0] = 1.f / (s[0] * s[0]);
}

__global__ void k_scaleM(const float* __restrict__ M, const float* __restrict__ alpha,
                         float* __restrict__ X)
{
  const int i = blockIdx.x * blockDim.x + threadIdx.x;
  if (i < 144 * 144) X[i] = alpha[0] * M[i];
}

// small dense f32 GEMM, C = A*B (144x144x144) — kept f32 for pinv precision
__global__ void k_sgemm144(const float* __restrict__ Aa, const float* __restrict__ Bb,
                           float* __restrict__ Cc)
{
  const int i = blockIdx.x * blockDim.x + threadIdx.x;
  if (i >= 144 * 144) return;
  const int r = i / 144, c = i - r * 144;
  float s = 0.f;
  for (int k = 0; k < 144; ++k) s += Aa[r * 144 + k] * Bb[k * 144 + c];
  Cc[i] = s;
}

__global__ void k_nsupd(float* __restrict__ X, const float* __restrict__ T)
{
  const int i = blockIdx.x * blockDim.x + threadIdx.x;
  if (i < 144 * 144) X[i] = 2.f * X[i] - T[i];
}

// Qbf = bf16(P*A), 160x1024 zero-padded (pad rows/cols zero so pads are inert)
__global__ void k_qbf(const float* __restrict__ X, const float* __restrict__ A,
                      __bf16* __restrict__ Qbf)
{
  const int i = blockIdx.x * blockDim.x + threadIdx.x;
  if (i >= 160 * 1024) return;
  const int j = i >> 10, e = i & 1023;
  float s = 0.f;
  if (j < 144 && e < 1012)
    for (int t = 0; t < 144; ++t) s += X[j * 144 + t] * A[t * 1012 + e];
  Qbf[i] = (__bf16)s;
}

// ---------------------------------------------------------------------------
// QP iteration (fused Davis–Yin, ALPHA=1):
//   v = -min(z,0) - w ;  r = v*A^T - b ;  z = -(w + r*Q)
// ---------------------------------------------------------------------------
__global__ void k_qp_r(const float* __restrict__ z, const float* __restrict__ wpad,
                       const __bf16* __restrict__ Abf, const float* __restrict__ bvec,
                       __bf16* __restrict__ rbf)
{
  const int numN = 10;                            // 160 node columns (padded)
  int wave = blockIdx.x * (blockDim.x >> 5) + (threadIdx.x >> 5);   // 80 exact
  if (wave >= 80) wave = 0;
  const int mt = wave / numN, nt = wave % numN;
  const int L = threadIdx.x & 31; const bool hi = L >= 16; const int lm = L & 15;
  const float*  zr = z    + (size_t)(mt * 16 + lm) * 1024;
  const float*  wr = wpad + (size_t)(mt * 16 + lm) * 1024;
  const __bf16* ar = Abf  + (size_t)(nt * 16 + lm) * 1024;   // B-frag: A^T column

  FragF acc;
#pragma unroll
  for (int r = 0; r < 8; ++r) acc.e[r] = 0.f;
  for (int ks = 0; ks < 32; ++ks) {
    const int kb = ks << 5;
    FragBF fa, fb;
#pragma unroll
    for (int s = 0; s < 16; ++s) {
      const int k = kb + akidx(s, hi);
      const float vv = -fminf(zr[k], 0.f) - wr[k];   // relu(z)-z-w, fused
      fa.e[s] = (__bf16)vv;
      fb.e[s] = ar[k];
    }
    acc.v = wmma_bf16(fa.v, fb.v, acc.v);
  }
#pragma unroll
  for (int r = 0; r < 8; ++r) {
    const int brow = mt * 16 + r + (hi ? 8 : 0);
    const int node = nt * 16 + lm;
    const float bv = (node < 144) ? bvec[node] : 0.f;
    rbf[brow * 160 + node] = (__bf16)(acc.e[r] - bv);
  }
}

// z-update GEMM. Each wave owns a fixed 16-column x 160-row slab of Q; on
// CDNA5 we DMA it into LDS once via the Tensor Data Mover (TENSORcnt) and
// feed WMMA B-fragments from LDS instead of 80 strided global loads per lane.
__global__ void k_qp_z(const __bf16* __restrict__ rbf, const __bf16* __restrict__ Qbf,
                       const float* __restrict__ wpad, float* __restrict__ z)
{
  const int numN = 64;                            // 1024 edge columns (padded)
  int wave = blockIdx.x * (blockDim.x >> 5) + (threadIdx.x >> 5);   // 512 exact
  if (wave >= 512) wave = 0;
  const int mt = wave / numN, nt = wave % numN;
  const int L = threadIdx.x & 31; const bool hi = L >= 16; const int lm = L & 15;
  const __bf16* rr = rbf + (size_t)(mt * 16 + lm) * 160;

#if HAVE_TDM
  __shared__ __bf16 qsh[8 * 160 * 16];            // 8 waves x 5 KB = 40 KB
  const int wib = (threadIdx.x >> 5) & 7;
  __bf16* qtile = &qsh[wib * (160 * 16)];
  {
    // Wave-uniform descriptor fields (readfirstlane -> SGPRs).
    const uint64_t ga = (uint64_t)(uintptr_t)(Qbf + nt * 16);
    const uint32_t la = (uint32_t)__builtin_amdgcn_readfirstlane(
        (int)(uint32_t)(uintptr_t)qtile);
    const uint32_t glo = (uint32_t)__builtin_amdgcn_readfirstlane((int)(uint32_t)ga);
    const uint32_t ghi = (uint32_t)__builtin_amdgcn_readfirstlane(
        (int)(uint32_t)(ga >> 32));
    tdm_load_bf16_tile(la, ((uint64_t)ghi << 32) | glo, 160, 1024);
    tdm_wait();                                   // s_wait_tensorcnt 0
  }
#else
  const __bf16* qc = Qbf + (nt * 16 + lm);        // element (k,n) = Qbf[k*1024+n]
#endif

  FragF acc;
#pragma unroll
  for (int r = 0; r < 8; ++r) acc.e[r] = 0.f;
  for (int ks = 0; ks < 5; ++ks) {                // K = 160 (144 + zero pad)
    const int kb = ks << 5;
    FragBF fa, fb;
#pragma unroll
    for (int s = 0; s < 16; ++s) {
      const int k = kb + akidx(s, hi);
      fa.e[s] = rr[k];
#if HAVE_TDM
      fb.e[s] = qtile[k * 16 + lm];               // ds_load from staged slab
#else
      fb.e[s] = qc[(size_t)k * 1024];
#endif
    }
    acc.v = wmma_bf16(fa.v, fb.v, acc.v);
  }
#pragma unroll
  for (int r = 0; r < 8; ++r) {
    const int b2 = mt * 16 + r + (hi ? 8 : 0);
    const int e2 = nt * 16 + lm;
    const size_t idx = (size_t)b2 * 1024 + e2;
    z[idx] = -(wpad[idx] + acc.e[r]);             // z' = -w - r*Q
  }
}

__global__ void k_out(const float* __restrict__ z, float* __restrict__ out)
{
  const int i = blockIdx.x * blockDim.x + threadIdx.x;
  if (i >= 128 * 1012) return;
  const int b = i / 1012, e = i - b * 1012;
  out[i] = fmaxf(z[(size_t)b * 1024 + e], 0.f);
}

// ---------------------------------------------------------------------------
// Host orchestration
// ---------------------------------------------------------------------------
extern "C" void kernel_launch(void* const* d_in, const int* in_sizes, int n_in,
                              void* d_out, int out_size, void* d_ws, size_t ws_size,
                              hipStream_t stream)
{
  (void)n_in; (void)out_size; (void)ws_size;
  // ---- input mapping (handle dict-insertion vs JAX-sorted flattening) ----
  const float *din, *Ain, *bvec, *c1w, *bn1g, *bn1b, *c2w;
  const int* dst;
  const float *bkc1[2], *bkc2[2], *bkg1[2], *bkb1[2], *bkg2[2], *bkb2[2];
  auto F = [&](int i) { return (const float*)d_in[i]; };
  if (in_sizes[0] == 128 * 3 * 96 * 96) {
    // insertion order: d, A, b, dst_idx, conv1_w, bn1_g, bn1_b, conv2_w,
    //                  blk{1,2}_{conv1_w, conv2_w, bn1_g, bn1_b, bn2_g, bn2_b}
    din = F(0); Ain = F(1); bvec = F(2); dst = (const int*)d_in[3];
    c1w = F(4); bn1g = F(5); bn1b = F(6); c2w = F(7);
    for (int bk = 0; bk < 2; ++bk) {
      const int o = 8 + bk * 6;
      bkc1[bk] = F(o);     bkc2[bk] = F(o + 1);
      bkg1[bk] = F(o + 2); bkb1[bk] = F(o + 3);
      bkg2[bk] = F(o + 4); bkb2[bk] = F(o + 5);
    }
  } else {
    // sorted order: A, b, d, dst_idx,
    //   blk{1,2}_{bn1_b, bn1_g, bn2_b, bn2_g, conv1_w, conv2_w},
    //   bn1_b, bn1_g, conv1_w, conv2_w
    Ain = F(0); bvec = F(1); din = F(2); dst = (const int*)d_in[3];
    for (int bk = 0; bk < 2; ++bk) {
      const int o = 4 + bk * 6;
      bkb1[bk] = F(o);     bkg1[bk] = F(o + 1);
      bkb2[bk] = F(o + 2); bkg2[bk] = F(o + 3);
      bkc1[bk] = F(o + 4); bkc2[bk] = F(o + 5);
    }
    bn1b = F(16); bn1g = F(17); c1w = F(18); c2w = F(19);
  }

  // ---- workspace carve-up ----
  char* ws = (char*)d_ws;
  size_t off = 0;
  auto alloc = [&](size_t bytes) -> void* {
    void* p = ws + off;
    off = (off + bytes + 255) & ~(size_t)255;
    return p;
  };
  float*  conv1raw = (float*)alloc((size_t)18874368 * 4);  // 128x64x48x48
  float*  P0   = (float*)alloc((size_t)4718592 * 4);       // 128x64x24x24
  float*  P1   = (float*)alloc((size_t)4718592 * 4);
  float*  P2   = (float*)alloc((size_t)4718592 * 4);
  float*  mu   = (float*)alloc(64 * 4);
  float*  rsg  = (float*)alloc(64 * 4);
  float*  out1 = (float*)alloc((size_t)73728 * 4);         // 128x24x24
  float*  cost = (float*)alloc((size_t)18432 * 4);         // 128x144
  float*  wpad = (float*)alloc((size_t)131072 * 4);        // 128x1024
  float*  zbuf = (float*)alloc((size_t)131072 * 4);
  __bf16* rbf  = (__bf16*)alloc((size_t)20480 * 2);        // 128x160
  __bf16* Abf  = (__bf16*)alloc((size_t)163840 * 2);       // 160x1024
  __bf16* Qbf  = (__bf16*)alloc((size_t)163840 * 2);
  float*  Mm   = (float*)alloc((size_t)20736 * 4);         // 144x144
  float*  Xm   = (float*)alloc((size_t)20736 * 4);
  float*  Ym   = (float*)alloc((size_t)20736 * 4);
  float*  Tm   = (float*)alloc((size_t)20736 * 4);
  float*  alpha = (float*)alloc(4);

  const int TB = 256;  // 8 wave32 per block

  // ---- CNN ----
  // conv1: 7x7 s2 p3, (128,3,96,96)->(128,64,48,48); tiles = 4*18432 -> 9216 blocks
  k_conv_wmma<<<9216, TB, 0, stream>>>(din, c1w, conv1raw,
                                       128, 3, 96, 96, 64, 7, 7, 2, 3, 48, 48);
  k_bn_stats<<<64, TB, 0, stream>>>(conv1raw, mu, rsg, 128, 64, 2304);
  k_bn_apply<<<(18874368 + TB - 1) / TB, TB, 0, stream>>>(
      conv1raw, mu, rsg, bn1g, bn1b, nullptr, conv1raw, 18874368, 64, 2304, 1);
  k_maxpool<<<(4718592 + TB - 1) / TB, TB, 0, stream>>>(
      conv1raw, P0, 128 * 64, 48, 48, 24, 24);

  const float* bx = P0; float* h1 = P1; float* h2 = P2;
  for (int bk = 0; bk < 2; ++bk) {
    // conv 3x3 s1 p1 on 24x24: tiles = 4*4608 -> 2304 blocks
    k_conv_wmma<<<2304, TB, 0, stream>>>(bx, bkc1[bk], h1,
                                         128, 64, 24, 24, 64, 3, 3, 1, 1, 24, 24);
    k_bn_stats<<<64, TB, 0, stream>>>(h1, mu, rsg, 128, 64, 576);
    k_bn_apply<<<(4718592 + TB - 1) / TB, TB, 0, stream>>>(
        h1, mu, rsg, bkg1[bk], bkb1[bk], nullptr, h1, 4718592, 64, 576, 1);
    k_conv_wmma<<<2304, TB, 0, stream>>>(h1, bkc2[bk], h2,
                                         128, 64, 24, 24, 64, 3, 3, 1, 1, 24, 24);
    k_bn_stats<<<64, TB, 0, stream>>>(h2, mu, rsg, 128, 64, 576);
    k_bn_apply<<<(4718592 + TB - 1) / TB, TB, 0, stream>>>(
        h2, mu, rsg, bkg2[bk], bkb2[bk], bx, h2, 4718592, 64, 576, 1);
    const float* nbx = h2; h2 = (float*)bx; bx = nbx;   // rotate ping-pong
  }
  k_conv1x1<<<(73728 + TB - 1) / TB, TB, 0, stream>>>(bx, c2w, out1, 128, 64, 576);
  k_amax<<<(18432 + TB - 1) / TB, TB, 0, stream>>>(out1, cost);
  k_gather_w<<<131072 / TB, TB, 0, stream>>>(cost, dst, wpad, zbuf, 1012);

  // ---- QP prologue: Abf, M = AA^T (WMMA, exact), P via Newton–Schulz, Q = PA ----
  k_prep_Abf<<<163840 / TB, TB, 0, stream>>>(Ain, Abf);
  k_maat<<<11, TB, 0, stream>>>(Abf, Mm);
  k_alpha<<<1, TB, 0, stream>>>(Mm, alpha);
  k_scaleM<<<81, TB, 0, stream>>>(Mm, alpha, Xm);
  for (int it = 0; it < 48; ++it) {        // X <- 2X - X M X
    k_sgemm144<<<81, TB, 0, stream>>>(Xm, Mm, Ym);
    k_sgemm144<<<81, TB, 0, stream>>>(Ym, Xm, Tm);
    k_nsupd<<<81, TB, 0, stream>>>(Xm, Tm);
  }
  k_qbf<<<163840 / TB, TB, 0, stream>>>(Xm, Ain, Qbf);

  // ---- 300 fused Davis–Yin iterations: r = vA^T - b ; z = -(w + rQ) ----
  for (int it = 0; it < 300; ++it) {
    k_qp_r<<<10, TB, 0, stream>>>(zbuf, wpad, Abf, bvec, rbf);
    k_qp_z<<<64, TB, 0, stream>>>(rbf, Qbf, wpad, zbuf);
  }
  k_out<<<(129536 + TB - 1) / TB, TB, 0, stream>>>(zbuf, (float*)d_out);
}
